// W8A16Linear_9096740733404
// MI455X (gfx1250) — matrix operand order repaired and checked by reading, not compiled, not run on hardware
//
#include <hip/hip_runtime.h>

// ---- problem constants (match reference) ----
#define M_TOT 16384   // B*S = 8*2048
#define N_TOT 16384   // DOUT
#define K_TOT 4096    // DIN

// ---- tiling ----
#define BM 128
#define BN 128
#define BK 32
#define NK (K_TOT / BK)        // 128 K-iterations
#define LDS_STRIDE 40          // elements (80 bytes): 64B row + 16B pad, 16B-aligned

typedef __attribute__((ext_vector_type(16))) __bf16 v16bf;
typedef __attribute__((ext_vector_type(8)))  __bf16 v8bf;
typedef __attribute__((ext_vector_type(8)))  float  v8f;

__global__ __launch_bounds__(256)
void w8a16_wmma_kernel(const __bf16* __restrict__ X,
                       const int*    __restrict__ W,
                       const __bf16* __restrict__ Scl,
                       __bf16*       __restrict__ Out)
{
    // double-buffered staging: 2 x (10 KB X + 10 KB W) = 40 KB LDS
    __shared__ __bf16 Xs[2][BM * LDS_STRIDE];
    __shared__ __bf16 Ws[2][BN * LDS_STRIDE];

    const int t    = threadIdx.x;
    const int lane = t & 31;
    const int wave = t >> 5;
    const int wm   = wave >> 2;   // 0..1 -> 64 output rows per wave
    const int wn   = wave & 3;    // 0..3 -> 32 output cols per wave

    const int bm = blockIdx.y * BM;
    const int bn = blockIdx.x * BN;

    // ---- X staging: each thread owns two 16B chunks of the 128x32 bf16 tile ----
    const int xc0 = t;            // chunk ids 0..511 (4 x 16B chunks per 32-elem row)
    const int xc1 = t + 256;
    const int xr0 = xc0 >> 2, xk0 = (xc0 & 3) * 8;
    const int xr1 = xc1 >> 2, xk1 = (xc1 & 3) * 8;
    const __bf16* xg0 = X + (size_t)(bm + xr0) * K_TOT + xk0;
    const __bf16* xg1 = X + (size_t)(bm + xr1) * K_TOT + xk1;

    // ---- W staging/dequant: each thread owns 16 int32 (half a 32-wide row) ----
    const int   wrow = t >> 1;            // n within tile: 0..127
    const int   wk   = (t & 1) * 16;      // k offset within tile
    const float scl  = (float)Scl[bn + wrow];            // hoisted per-channel scale
    const int*  wg   = W + (size_t)(bn + wrow) * K_TOT + wk;

    v8f acc[4][2];
    #pragma unroll
    for (int i = 0; i < 4; ++i)
        #pragma unroll
        for (int j = 0; j < 2; ++j)
            #pragma unroll
            for (int e = 0; e < 8; ++e)
                acc[i][j][e] = 0.0f;

    const int half = (lane < 16) ? 0 : 1;   // which K half-group this lane holds
    const int l16  = lane & 15;

    // ---------- prologue: stage tile 0 ----------
    {
        uint4 xa = *(const uint4*)xg0;
        uint4 xb = *(const uint4*)xg1;
        const int4* wp = (const int4*)wg;
        int4 a0 = wp[0], a1 = wp[1], a2 = wp[2], a3 = wp[3];

        *(uint4*)&Xs[0][xr0 * LDS_STRIDE + xk0] = xa;
        *(uint4*)&Xs[0][xr1 * LDS_STRIDE + xk1] = xb;

        int v[16] = { a0.x, a0.y, a0.z, a0.w, a1.x, a1.y, a1.z, a1.w,
                      a2.x, a2.y, a2.z, a2.w, a3.x, a3.y, a3.z, a3.w };
        v8bf h0, h1;
        #pragma unroll
        for (int i = 0; i < 8; ++i) {
            h0[i] = (__bf16)((float)v[i]     * scl);
            h1[i] = (__bf16)((float)v[i + 8] * scl);
        }
        *(v8bf*)&Ws[0][wrow * LDS_STRIDE + wk]     = h0;
        *(v8bf*)&Ws[0][wrow * LDS_STRIDE + wk + 8] = h1;
    }
    __syncthreads();

    // ---------- pipelined main loop: one barrier per iteration ----------
    for (int it = 0; it < NK; ++it) {
        const int  cur  = it & 1;
        const bool more = (it + 1) < NK;

        // 1) issue global loads for tile it+1 (LOADcnt in flight during WMMA)
        uint4 nxa, nxb;
        int4  nw0, nw1, nw2, nw3;
        if (more) {
            const int k = (it + 1) * BK;
            nxa = *(const uint4*)(xg0 + k);
            nxb = *(const uint4*)(xg1 + k);
            const int4* wp = (const int4*)(wg + k);
            nw0 = wp[0]; nw1 = wp[1]; nw2 = wp[2]; nw3 = wp[3];
        }

        // 2) fragments in ISA wave32 WMMA layout + 8 WMMAs on tile `it`
        // A (16x32 bf16): lane M=l16; half 0 holds K {0..7,16..23}, half 1 {8..15,24..31}
        v16bf afr[4];
        #pragma unroll
        for (int tm = 0; tm < 4; ++tm) {
            const int row = wm * 64 + tm * 16 + l16;
            const __bf16* p = &Xs[cur][row * LDS_STRIDE + half * 8];
            v8bf lo = *(const v8bf*)p;
            v8bf hi = *(const v8bf*)(p + 16);
            afr[tm] = __builtin_shufflevector(lo, hi,
                        0,1,2,3,4,5,6,7,8,9,10,11,12,13,14,15);
        }
        // B (32x16 bf16): lane N=l16; half 0 holds K {0..7,16..23}, half 1 {8..15,24..31}
        v16bf bfr[2];
        #pragma unroll
        for (int tn = 0; tn < 2; ++tn) {
            const int row = wn * 32 + tn * 16 + l16;
            const __bf16* p = &Ws[cur][row * LDS_STRIDE + half * 8];
            v8bf lo = *(const v8bf*)p;
            v8bf hi = *(const v8bf*)(p + 16);
            bfr[tn] = __builtin_shufflevector(lo, hi,
                        0,1,2,3,4,5,6,7,8,9,10,11,12,13,14,15);
        }

        #pragma unroll
        for (int tm = 0; tm < 4; ++tm)
            #pragma unroll
            for (int tn = 0; tn < 2; ++tn)
                acc[tm][tn] = __builtin_amdgcn_wmma_f32_16x16x32_bf16(
                    /*neg_a=*/false, afr[tm],
                    /*neg_b=*/false, bfr[tn],
                    /*c_mod=*/(short)0, acc[tm][tn],
                    /*reuse_a=*/false, /*reuse_b=*/false);

        // 3) dequantize + stage tile it+1 into the other buffer
        if (more) {
            const int nxt = cur ^ 1;
            *(uint4*)&Xs[nxt][xr0 * LDS_STRIDE + xk0] = nxa;
            *(uint4*)&Xs[nxt][xr1 * LDS_STRIDE + xk1] = nxb;

            int v[16] = { nw0.x, nw0.y, nw0.z, nw0.w, nw1.x, nw1.y, nw1.z, nw1.w,
                          nw2.x, nw2.y, nw2.z, nw2.w, nw3.x, nw3.y, nw3.z, nw3.w };
            v8bf h0, h1;
            #pragma unroll
            for (int i = 0; i < 8; ++i) {
                h0[i] = (__bf16)((float)v[i]     * scl);
                h1[i] = (__bf16)((float)v[i + 8] * scl);
            }
            *(v8bf*)&Ws[nxt][wrow * LDS_STRIDE + wk]     = h0;
            *(v8bf*)&Ws[nxt][wrow * LDS_STRIDE + wk + 8] = h1;
        }

        __syncthreads();
    }

    // ---- epilogue: f32 acc -> bf16; D layout: VGPR i = row (i + half*8), lane%16 = col ----
    #pragma unroll
    for (int tm = 0; tm < 4; ++tm) {
        #pragma unroll
        for (int tn = 0; tn < 2; ++tn) {
            const int mbase = bm + wm * 64 + tm * 16 + half * 8;
            const int n     = bn + wn * 32 + tn * 16 + l16;
            #pragma unroll
            for (int i = 0; i < 8; ++i) {
                Out[(size_t)(mbase + i) * N_TOT + n] = (__bf16)acc[tm][tn][i];
            }
        }
    }
}

extern "C" void kernel_launch(void* const* d_in, const int* in_sizes, int n_in,
                              void* d_out, int out_size, void* d_ws, size_t ws_size,
                              hipStream_t stream) {
    const __bf16* X = (const __bf16*)d_in[0];   // input_tensor bf16 [B,S,DIN]
    const int*    W = (const int*)d_in[1];      // weight int32 [DOUT,DIN]
    const __bf16* S = (const __bf16*)d_in[2];   // weight_scale bf16 [DOUT,1]
    __bf16*       O = (__bf16*)d_out;           // output bf16 [B,S,DOUT]

    dim3 grid(N_TOT / BN, M_TOT / BM);          // (128, 128)
    w8a16_wmma_kernel<<<grid, 256, 0, stream>>>(X, W, S, O);
}
